// SpatialSelfAttention_52647709115054
// MI455X (gfx1250) — compile-verified, matrix operand order checked by reading, not run on hardware
//
#include <hip/hip_runtime.h>
#include <hip/hip_bf16.h>

typedef __attribute__((ext_vector_type(16))) _Float16     v16h;
typedef __attribute__((ext_vector_type(8)))  float        v8f;
typedef __attribute__((ext_vector_type(4)))  unsigned int v4u;
typedef __attribute__((ext_vector_type(8)))  int          v8i;
typedef __attribute__((ext_vector_type(4)))  int          v4i;

#define CDIM 512
#define NTOK 4096
#define GRPS 32

#if __has_builtin(__builtin_amdgcn_tensor_load_to_lds)
#define HAVE_TDM 1
#else
#define HAVE_TDM 0
#endif

static __device__ __forceinline__ v8f wmma_f16(v16h a, v16h b, v8f c) {
  // v_wmma_f32_16x16x32_f16  D = A(16x32) * B(32x16) + C(16x16 f32)
  return __builtin_amdgcn_wmma_f32_16x16x32_f16(
      /*neg_a=*/false, a, /*neg_b=*/false, b,
      /*c_mod=*/(short)0, c, /*reuse_a=*/false, /*reuse_b=*/false);
}

#if HAVE_TDM
// Issue a TDM 2D tile load: tensor of `dim0 x dim1` elements (2B each,
// row stride = stride0 elements), tile of `t0 x t1` elements starting at
// `gaddr`, landing at LDS byte offset `lds_addr`.  D# packing per
// CDNA5 ISA sec.8 (group0: count=1, type=2 "image").
static __device__ __forceinline__ void tdm_load_2d(unsigned lds_addr,
                                                   unsigned long long gaddr,
                                                   unsigned dim0, unsigned dim1,
                                                   unsigned t0, unsigned t1,
                                                   unsigned stride0) {
  v4u g0;
  g0.x = 1u;                                            // count=1, user mode
  g0.y = lds_addr;                                      // lds_addr (bytes)
  g0.z = (unsigned)gaddr;                               // global_addr[31:0]
  g0.w = ((unsigned)(gaddr >> 32) & 0x1FFFFFFu) | (2u << 30);  // addr[56:32]|type=2
  v8i g1;
  g1[0] = (int)(1u << 16);              // workgroup_mask=0, data_size=1 (2B)
  g1[1] = (int)(dim0 << 16);            // tensor_dim0[15:0]  -> bits 63:48
  g1[2] = (int)((dim0 >> 16) | (dim1 << 16));   // dim0[31:16] | dim1[15:0]
  g1[3] = (int)((dim1 >> 16) | (t0 << 16));     // dim1[31:16] | tile_dim0
  g1[4] = (int)(t1);                    // tile_dim1 | tile_dim2=0
  g1[5] = (int)stride0;                 // tensor_dim0_stride[31:0]
  g1[6] = 0;                            // stride0 hi | stride1 lo
  g1[7] = 0;                            // stride1 hi
  v4i z = {0, 0, 0, 0};
#if defined(__clang_major__) && (__clang_major__ >= 23)
  v8i z8 = {0, 0, 0, 0, 0, 0, 0, 0};
  __builtin_amdgcn_tensor_load_to_lds(g0, g1, z, z, z8, 0);
#else
  __builtin_amdgcn_tensor_load_to_lds(g0, g1, z, z, 0);
#endif
}
#endif

// ---------------------------------------------------------------------------
// Kernel 1: GroupNorm(32, eps=1e-6) -> fp16 tokens h16[b][n][c]
// ---------------------------------------------------------------------------
__global__ void gn_kernel(const float* __restrict__ x,
                          const float* __restrict__ gw,
                          const float* __restrict__ gb,
                          _Float16* __restrict__ h16) {
  const int CPG = CDIM / GRPS;            // 16 channels per group
  const int b   = blockIdx.x >> 5;
  const int grp = blockIdx.x & 31;
  const int c0  = grp * CPG;
  const int tid = threadIdx.x;
  const float* xb = x + ((size_t)b * CDIM + c0) * NTOK;

  float s = 0.f, ss = 0.f;
  for (int i = tid; i < CPG * NTOK; i += 256) {
    float v = xb[i];
    s += v; ss += v * v;
  }
  __shared__ float red[256], red2[256];
  red[tid] = s; red2[tid] = ss;
  __syncthreads();
  for (int o = 128; o > 0; o >>= 1) {
    if (tid < o) { red[tid] += red[tid + o]; red2[tid] += red2[tid + o]; }
    __syncthreads();
  }
  const float inv_n = 1.0f / (float)(CPG * NTOK);
  float mu   = red[0] * inv_n;
  float var  = red2[0] * inv_n - mu * mu;
  float rstd = rsqrtf(var + 1e-6f);

  for (int i = tid; i < CPG * NTOK; i += 256) {
    int c = c0 + (i >> 12);
    int n = i & (NTOK - 1);
    float v = (xb[i] - mu) * rstd * gw[c] + gb[c];
    h16[((size_t)b * NTOK + n) * CDIM + c] = (_Float16)v;   // token-major
  }
}

// ---------------------------------------------------------------------------
// Kernel 2: WMMA GEMM  D[o][n] = W[o][c] * h16^T[c][n] + bias
// Block tile 64(m) x 128(n), BK=32, 128 threads (4 waves), wave = 2x4 tiles.
//   mode 0/1: q/k -> out[n][c] token-major (q pre-scaled by C^-1/2)
//   mode 2:   v   -> out[c][n] channel-major
// ---------------------------------------------------------------------------
__global__ void qkv_gemm_kernel(const _Float16* __restrict__ h16,
                                const float* __restrict__ W,
                                const float* __restrict__ bias,
                                _Float16* __restrict__ out,
                                int mode, float scale) {
  __shared__ __attribute__((aligned(32))) _Float16 As[64][32];   // [m][k]
  __shared__ __attribute__((aligned(32))) _Float16 Bs[128][32];  // [n][k]

  const int b  = blockIdx.z;
  const int n0 = blockIdx.x * 128;
  const int m0 = blockIdx.y * 64;
  const int tid = threadIdx.x;
  const int lane = tid & 31, w = tid >> 5;
  const int wr = w >> 1, wc = w & 1;
  const int g = lane >> 4, lr = lane & 15;

  const _Float16* Bg = h16 + (size_t)b * NTOK * CDIM;

  v8f acc[2][4] = {};

  for (int kk = 0; kk < CDIM; kk += 32) {
    __syncthreads();
    {   // stage A: fp32 weights -> fp16, 64x32 tile
      int r = tid >> 1, seg = tid & 1;
      const float4* src = (const float4*)(W + (size_t)(m0 + r) * CDIM + kk + seg * 16);
      #pragma unroll
      for (int q4 = 0; q4 < 4; ++q4) {
        float4 f = src[q4];
        As[r][seg * 16 + q4 * 4 + 0] = (_Float16)f.x;
        As[r][seg * 16 + q4 * 4 + 1] = (_Float16)f.y;
        As[r][seg * 16 + q4 * 4 + 2] = (_Float16)f.z;
        As[r][seg * 16 + q4 * 4 + 3] = (_Float16)f.w;
      }
    }
    {   // stage B: fp16 tokens, 128x32 tile, one 64B row per thread
      const uint4* src = (const uint4*)(Bg + (size_t)(n0 + tid) * CDIM + kk);
      uint4* dst = (uint4*)(&Bs[tid][0]);
      dst[0] = src[0]; dst[1] = src[1]; dst[2] = src[2]; dst[3] = src[3];
    }
    __syncthreads();
    if (kk + 32 < CDIM) {   // keep next K tile warm at WGP scope
      __builtin_prefetch(W  + (size_t)(m0 + (tid >> 1)) * CDIM + kk + 32, 0, 3);
      __builtin_prefetch(Bg + (size_t)(n0 + tid) * CDIM + kk + 32, 0, 3);
    }

    v16h a[2], bb[4];
    #pragma unroll
    for (int i = 0; i < 2; ++i)
      a[i]  = *(const v16h*)(&As[wr * 32 + i * 16 + lr][g * 16]);
    #pragma unroll
    for (int j = 0; j < 4; ++j)
      bb[j] = *(const v16h*)(&Bs[wc * 64 + j * 16 + lr][g * 16]);
    #pragma unroll
    for (int i = 0; i < 2; ++i)
      #pragma unroll
      for (int j = 0; j < 4; ++j)
        acc[i][j] = wmma_f16(a[i], bb[j], acc[i][j]);
  }

  #pragma unroll
  for (int i = 0; i < 2; ++i) {
    #pragma unroll
    for (int j = 0; j < 4; ++j) {
      #pragma unroll
      for (int v = 0; v < 8; ++v) {
        int row = m0 + wr * 32 + i * 16 + v + 8 * g;   // output channel o
        int col = n0 + wc * 64 + j * 16 + lr;          // token n
        float val = (acc[i][j][v] + bias[row]) * scale;
        if (mode == 2)
          out[(size_t)b * CDIM * NTOK + (size_t)row * NTOK + col] = (_Float16)val;
        else
          out[(size_t)b * NTOK * CDIM + (size_t)col * CDIM + row] = (_Float16)val;
      }
    }
  }
}

// ---------------------------------------------------------------------------
// Kernel 3: flash attention. 16 queries/block, 128 threads (4 waves).
// Wave w owns V-dims [w*128, w*128+128). Key tiles of 32, online softmax.
// K/V tiles staged via the Tensor Data Mover when available.
// LDS: Qs 16KB + Ks 32KB + Vs 32KB + Ps 4KB = 84KB (<320KB/WGP).
// ---------------------------------------------------------------------------
__global__ void attn_kernel(const _Float16* __restrict__ q16,   // [b][n][c]
                            const _Float16* __restrict__ k16,   // [b][n][c]
                            const _Float16* __restrict__ v16,   // [b][c][n]
                            _Float16* __restrict__ o16) {       // [b][n][c]
  __shared__ __attribute__((aligned(32))) _Float16 Qs[16][CDIM];
  __shared__ __attribute__((aligned(32))) _Float16 Ks[32][CDIM];
  __shared__ __attribute__((aligned(32))) _Float16 Vs[CDIM][32];
  __shared__ __attribute__((aligned(32))) _Float16 Ps[4][16][32];

  const int b  = blockIdx.y;
  const int n0 = blockIdx.x * 16;
  const int tid = threadIdx.x;
  const int lane = tid & 31, w = tid >> 5;
  const int g = lane >> 4, lr = lane & 15;

  const _Float16* Qg = q16 + (size_t)b * NTOK * CDIM;
  const _Float16* Kg = k16 + (size_t)b * NTOK * CDIM;
  const _Float16* Vg = v16 + (size_t)b * CDIM * NTOK;

  {   // stage Q once: 16 rows x 512 halves, 128B per thread
    int r = tid >> 3, seg = tid & 7;
    const uint4* src = (const uint4*)(Qg + (size_t)(n0 + r) * CDIM + seg * 64);
    uint4* dst = (uint4*)(&Qs[r][seg * 64]);
    #pragma unroll
    for (int i = 0; i < 8; ++i) dst[i] = src[i];
  }

  v8f acc[8] = {};
  float m_run[8], l_run[8];
  #pragma unroll
  for (int v = 0; v < 8; ++v) { m_run[v] = -1e30f; l_run[v] = 0.f; }

  for (int kt = 0; kt < NTOK / 32; ++kt) {
    const int k0 = kt * 32;
    __syncthreads();
#if HAVE_TDM
    if (tid < 32) {
      // K tile: tensor 512x4096 (row=token, 512 elems/row), tile 512x32 rows
      tdm_load_2d((unsigned)(size_t)&Ks[0][0],
                  (unsigned long long)(size_t)(Kg + (size_t)k0 * CDIM),
                  /*dim0=*/CDIM, /*dim1=*/NTOK, /*t0=*/CDIM, /*t1=*/32,
                  /*stride0=*/CDIM);
    } else if (tid < 64) {
      // V tile: tensor 4096x512 (row=channel), tile 32x512 rows
      tdm_load_2d((unsigned)(size_t)&Vs[0][0],
                  (unsigned long long)(size_t)(Vg + k0),
                  /*dim0=*/NTOK, /*dim1=*/CDIM, /*t0=*/32, /*t1=*/CDIM,
                  /*stride0=*/NTOK);
    }
    __builtin_amdgcn_s_wait_tensorcnt(0);
#else
    {   // stage K tile: 32 keys x 512 (token-major rows)
      int r = tid >> 2, seg = tid & 3;
      const uint4* src = (const uint4*)(Kg + (size_t)(k0 + r) * CDIM + seg * 128);
      uint4* dst = (uint4*)(&Ks[r][seg * 128]);
      #pragma unroll
      for (int i = 0; i < 16; ++i) dst[i] = src[i];
    }
    {   // stage V tile: 512 channel-rows x 32 keys (64B rows)
      #pragma unroll
      for (int rr = 0; rr < 4; ++rr) {
        int c = tid * 4 + rr;
        const uint4* src = (const uint4*)(Vg + (size_t)c * NTOK + k0);
        uint4* dst = (uint4*)(&Vs[c][0]);
        dst[0] = src[0]; dst[1] = src[1]; dst[2] = src[2]; dst[3] = src[3];
      }
    }
#endif
    __syncthreads();

    // S = Q * K^T : two 16x16 score tiles (keys 0-15, 16-31)
    v8f s0 = {}, s1 = {};
    for (int ck = 0; ck < CDIM; ck += 32) {
      v16h a  = *(const v16h*)(&Qs[lr][ck + g * 16]);
      v16h b0 = *(const v16h*)(&Ks[lr][ck + g * 16]);
      v16h b1 = *(const v16h*)(&Ks[16 + lr][ck + g * 16]);
      s0 = wmma_f16(a, b0, s0);
      s1 = wmma_f16(a, b1, s1);
    }

    // online softmax; lane-group g owns rows v + 8g of the 16x16 C/D layout
    float sf[8];
    #pragma unroll
    for (int v = 0; v < 8; ++v) {
      float mx = fmaxf(s0[v], s1[v]);
      #pragma unroll
      for (int off = 1; off < 16; off <<= 1)
        mx = fmaxf(mx, __shfl_xor(mx, off, 32));
      float mn = fmaxf(m_run[v], mx);
      float f  = __expf(m_run[v] - mn);
      float p0 = __expf(s0[v] - mn);
      float p1 = __expf(s1[v] - mn);
      float rs = p0 + p1;
      #pragma unroll
      for (int off = 1; off < 16; off <<= 1)
        rs += __shfl_xor(rs, off, 32);
      m_run[v] = mn;
      l_run[v] = l_run[v] * f + rs;
      sf[v] = f;
      Ps[w][v + 8 * g][lr]      = (_Float16)p0;
      Ps[w][v + 8 * g][16 + lr] = (_Float16)p1;
    }
    #pragma unroll
    for (int ct = 0; ct < 8; ++ct)
      #pragma unroll
      for (int v = 0; v < 8; ++v)
        acc[ct][v] *= sf[v];
    __syncthreads();

    // O += P * V for this wave's 128 V-dims (8 col tiles)
    v16h pf = *(const v16h*)(&Ps[w][lr][g * 16]);
    #pragma unroll
    for (int ct = 0; ct < 8; ++ct) {
      v16h bv = *(const v16h*)(&Vs[w * 128 + ct * 16 + lr][g * 16]);
      acc[ct] = wmma_f16(pf, bv, acc[ct]);
    }
  }

  _Float16* Og = o16 + (size_t)b * NTOK * CDIM;
  #pragma unroll
  for (int v = 0; v < 8; ++v) {
    float inv = 1.0f / l_run[v];
    int row = n0 + v + 8 * g;
    #pragma unroll
    for (int ct = 0; ct < 8; ++ct)
      Og[(size_t)row * CDIM + w * 128 + ct * 16 + lr] = (_Float16)(acc[ct][v] * inv);
  }
}

// ---------------------------------------------------------------------------
// Kernel 4: output projection + bias + residual, fp32 out[b][c][n]
// ---------------------------------------------------------------------------
__global__ void proj_gemm_kernel(const _Float16* __restrict__ o16,  // [b][n][c]
                                 const float* __restrict__ W,
                                 const float* __restrict__ bias,
                                 const float* __restrict__ x,
                                 float* __restrict__ out) {
  __shared__ __attribute__((aligned(32))) _Float16 As[64][32];
  __shared__ __attribute__((aligned(32))) _Float16 Bs[128][32];

  const int b  = blockIdx.z;
  const int n0 = blockIdx.x * 128;
  const int m0 = blockIdx.y * 64;
  const int tid = threadIdx.x;
  const int lane = tid & 31, w = tid >> 5;
  const int wr = w >> 1, wc = w & 1;
  const int g = lane >> 4, lr = lane & 15;

  const _Float16* Bg = o16 + (size_t)b * NTOK * CDIM;

  v8f acc[2][4] = {};

  for (int kk = 0; kk < CDIM; kk += 32) {
    __syncthreads();
    {
      int r = tid >> 1, seg = tid & 1;
      const float4* src = (const float4*)(W + (size_t)(m0 + r) * CDIM + kk + seg * 16);
      #pragma unroll
      for (int q4 = 0; q4 < 4; ++q4) {
        float4 f = src[q4];
        As[r][seg * 16 + q4 * 4 + 0] = (_Float16)f.x;
        As[r][seg * 16 + q4 * 4 + 1] = (_Float16)f.y;
        As[r][seg * 16 + q4 * 4 + 2] = (_Float16)f.z;
        As[r][seg * 16 + q4 * 4 + 3] = (_Float16)f.w;
      }
    }
    {
      const uint4* src = (const uint4*)(Bg + (size_t)(n0 + tid) * CDIM + kk);
      uint4* dst = (uint4*)(&Bs[tid][0]);
      dst[0] = src[0]; dst[1] = src[1]; dst[2] = src[2]; dst[3] = src[3];
    }
    __syncthreads();
    if (kk + 32 < CDIM) {
      __builtin_prefetch(W  + (size_t)(m0 + (tid >> 1)) * CDIM + kk + 32, 0, 3);
      __builtin_prefetch(Bg + (size_t)(n0 + tid) * CDIM + kk + 32, 0, 3);
    }

    v16h a[2], bb[4];
    #pragma unroll
    for (int i = 0; i < 2; ++i)
      a[i]  = *(const v16h*)(&As[wr * 32 + i * 16 + lr][g * 16]);
    #pragma unroll
    for (int j = 0; j < 4; ++j)
      bb[j] = *(const v16h*)(&Bs[wc * 64 + j * 16 + lr][g * 16]);
    #pragma unroll
    for (int i = 0; i < 2; ++i)
      #pragma unroll
      for (int j = 0; j < 4; ++j)
        acc[i][j] = wmma_f16(a[i], bb[j], acc[i][j]);
  }

  #pragma unroll
  for (int i = 0; i < 2; ++i) {
    #pragma unroll
    for (int j = 0; j < 4; ++j) {
      #pragma unroll
      for (int v = 0; v < 8; ++v) {
        int row = m0 + wr * 32 + i * 16 + v + 8 * g;   // channel o
        int col = n0 + wc * 64 + j * 16 + lr;          // spatial n
        size_t idx = (size_t)b * CDIM * NTOK + (size_t)row * NTOK + col;
        out[idx] = x[idx] + acc[i][j][v] + bias[row];
      }
    }
  }
}

// ---------------------------------------------------------------------------
extern "C" void kernel_launch(void* const* d_in, const int* in_sizes, int n_in,
                              void* d_out, int out_size, void* d_ws, size_t ws_size,
                              hipStream_t stream) {
  const float* x    = (const float*)d_in[0];
  const float* gn_w = (const float*)d_in[1];
  const float* gn_b = (const float*)d_in[2];
  const float* wq   = (const float*)d_in[3];
  const float* bq   = (const float*)d_in[4];
  const float* wk   = (const float*)d_in[5];
  const float* bk   = (const float*)d_in[6];
  const float* wv   = (const float*)d_in[7];
  const float* bv   = (const float*)d_in[8];
  const float* wo   = (const float*)d_in[9];
  const float* bo   = (const float*)d_in[10];
  float* out = (float*)d_out;

  char* ws = (char*)d_ws;
  const size_t seg = (size_t)2 * NTOK * CDIM * 2;   // 8MB of fp16 per tensor
  _Float16* h16 = (_Float16*)(ws + 0 * seg);
  _Float16* q16 = (_Float16*)(ws + 1 * seg);
  _Float16* k16 = (_Float16*)(ws + 2 * seg);
  _Float16* v16 = (_Float16*)(ws + 3 * seg);
  _Float16* o16 = (_Float16*)(ws + 4 * seg);

  const float scale = 0.044194173824159220f;   // 512^-0.5

  gn_kernel<<<2 * GRPS, 256, 0, stream>>>(x, gn_w, gn_b, h16);

  dim3 gg(NTOK / 128, CDIM / 64, 2);
  qkv_gemm_kernel<<<gg, 128, 0, stream>>>(h16, wq, bq, q16, /*mode=*/0, scale);
  qkv_gemm_kernel<<<gg, 128, 0, stream>>>(h16, wk, bk, k16, /*mode=*/1, 1.0f);
  qkv_gemm_kernel<<<gg, 128, 0, stream>>>(h16, wv, bv, v16, /*mode=*/2, 1.0f);

  attn_kernel<<<dim3(NTOK / 16, 2), 128, 0, stream>>>(q16, k16, v16, o16);

  proj_gemm_kernel<<<gg, 128, 0, stream>>>(o16, wo, bo, x, out);
}
